// X_vector_10496900072025
// MI455X (gfx1250) — matrix operand order repaired and checked by reading, not compiled, not run on hardware
//
#include <hip/hip_runtime.h>
#include <stdint.h>

// ---------------------------------------------------------------------------
// Types for CDNA5 WMMA (wave32): bf16 A/B fragments (8 VGPRs = 16 bf16/lane),
// f32 C/D accumulator (8 VGPRs).
// ---------------------------------------------------------------------------
typedef __attribute__((ext_vector_type(16))) __bf16 v16bf;
typedef __attribute__((ext_vector_type(8)))  float  v8f;

struct U4 { uint32_t a, b, c, d; };          // one 16-byte load
union Frag { v16bf v; U4 q[2]; };            // 32 bytes = 8 VGPRs

__device__ __forceinline__ float sigf(float x) { return 1.0f / (1.0f + __expf(-x)); }

// ---------------------------------------------------------------------------
// Pad input (64,1000,40) f32 -> (64,1000,64) bf16, channels 40..63 zero.
// ---------------------------------------------------------------------------
__global__ void pad_input_kernel(const float* __restrict__ in, __bf16* __restrict__ out,
                                 int total) {
  int i = blockIdx.x * 256 + threadIdx.x;
  if (i >= total) return;
  int c = i & 63;
  int bt = i >> 6;
  float v = (c < 40) ? in[bt * 40 + c] : 0.0f;
  out[i] = (__bf16)v;
}

// ---------------------------------------------------------------------------
// Weight convert: w (cs,Cin,512) f32 -> Wt (512, K=cs*CinPad) bf16, row-major
// over k so B-fragment loads are contiguous. Pads Cin -> CinPad with zeros.
// ---------------------------------------------------------------------------
__global__ void wconv_kernel(const float* __restrict__ w, __bf16* __restrict__ wt,
                             int Cin, int CinPad, int K) {
  int i = blockIdx.x * 256 + threadIdx.x;
  if (i >= 512 * K) return;
  int n = i / K;
  int k = i - n * K;
  int c = k / CinPad;
  int ci = k - c * CinPad;
  float v = (ci < Cin) ? w[(c * Cin + ci) * 512 + n] : 0.0f;
  wt[(long)n * K + k] = (__bf16)v;
}

// ---------------------------------------------------------------------------
// TDNN layer as WMMA GEMM.
//   out[b,t,n] = bn_scale[n]*relu(sum_k A[b,t,k]*Wt[n,k] + bias[n]) + bn_shift[n]
// A element k=(c,ci): act[b, t + c*dil, ci], Cin = 1<<cinShift (64 or 512).
// Block = 128 threads (4 waves); wave tile = 32(M) x 64(N); block = 128 x 64.
// grid = (64*Tout/128, 512/64). M divisible by 128 for all layers -> EXEC all 1s.
// ---------------------------------------------------------------------------
__global__ __launch_bounds__(128) void tdnn_wmma_kernel(
    const __bf16* __restrict__ act, const __bf16* __restrict__ wt,
    const float* __restrict__ bias, const float* __restrict__ bn_g,
    const float* __restrict__ bn_b, const float* __restrict__ bn_m,
    const float* __restrict__ bn_v, __bf16* __restrict__ out,
    int Tout, int cinShift, int K, int dil) {
  const int lane  = threadIdx.x & 31;
  const int wave  = threadIdx.x >> 5;
  const int laneM = lane & 15;   // A: row in tile / B,C,D: column in tile
  const int laneH = lane >> 4;   // K/M half selector per ISA layouts
  const int m0 = blockIdx.x * 128 + wave * 32;
  const int n0 = blockIdx.y * 64;

  // This lane's two A rows (mt=0,1): flat row r -> (b, t) in the T=1000 buffer.
  long rowbase[2];
#pragma unroll
  for (int mt = 0; mt < 2; ++mt) {
    int r = m0 + mt * 16 + laneM;
    int b = r / Tout;
    int t = r - b * Tout;
    rowbase[mt] = (long)b * 1000 + t;
  }

  const __bf16* wrow = wt + (long)(n0 + laneM) * K + laneH * 16;
  const long wstep = (long)16 * K;

  v8f acc[2][4] = {};

  for (int kk = 0; kk < K; kk += 32) {
    int c  = kk >> cinShift;
    int ci = kk - (c << cinShift);
    // A fragments: lane half 0 holds K {0..15}, half 1 holds {16..31}
    Frag A[2];
#pragma unroll
    for (int mt = 0; mt < 2; ++mt) {
      const __bf16* ap =
          act + (((rowbase[mt] + (long)c * dil) << cinShift) + ci) + laneH * 16;
      A[mt].q[0] = *(const U4*)(ap);
      A[mt].q[1] = *(const U4*)(ap + 8);
    }
    // B fragments: lane = column n, elements = K kk+laneH*16 .. +15 (contiguous)
    Frag B[4];
#pragma unroll
    for (int nt = 0; nt < 4; ++nt) {
      const __bf16* bp = wrow + nt * wstep + kk;
      B[nt].q[0] = *(const U4*)(bp);
      B[nt].q[1] = *(const U4*)(bp + 8);
    }
#pragma unroll
    for (int mt = 0; mt < 2; ++mt)
#pragma unroll
      for (int nt = 0; nt < 4; ++nt)
        acc[mt][nt] = __builtin_amdgcn_wmma_f32_16x16x32_bf16(
            false, A[mt].v, false, B[nt].v, (short)0, acc[mt][nt], false, false);
  }

  // Epilogue: C/D layout -> lane holds column n, rows m0+mt*16+laneH*8+j.
#pragma unroll
  for (int mt = 0; mt < 2; ++mt) {
    long orow[8];
#pragma unroll
    for (int j = 0; j < 8; ++j) {
      int r = m0 + mt * 16 + laneH * 8 + j;
      int b = r / Tout;
      int t = r - b * Tout;
      orow[j] = ((long)b * 1000 + t) * 512;
    }
#pragma unroll
    for (int nt = 0; nt < 4; ++nt) {
      int n = n0 + nt * 16 + laneM;
      float sc = bn_g[n] * rsqrtf(bn_v[n] + 1e-5f);
      float sh = bn_b[n] - bn_m[n] * sc;
      float bc = bias[n];
#pragma unroll
      for (int j = 0; j < 8; ++j) {
        float z = acc[mt][nt][j] + bc;
        z = fmaxf(z, 0.0f);
        z = z * sc + sh;
        out[orow[j] + n] = (__bf16)z;
      }
    }
  }
}

// ---------------------------------------------------------------------------
// Stats pooling: mean + unbiased var over T frames per (b, channel).
// block = 512 threads (one channel each), grid = 64 (batch).
// ---------------------------------------------------------------------------
__global__ __launch_bounds__(512) void stats_kernel(const __bf16* __restrict__ act,
                                                    float* __restrict__ stat, int T) {
  int b = blockIdx.x;
  int c = threadIdx.x;
  const __bf16* p = act + (long)b * 1000 * 512 + c;
  float s = 0.0f, sq = 0.0f;
  for (int t = 0; t < T; ++t) {
    float x = (float)p[(long)t * 512];
    s += x;
    sq += x * x;
  }
  float mean = s / (float)T;
  float var = (sq - s * s / (float)T) / (float)(T - 1);
  stat[b * 1024 + c] = mean;
  stat[b * 1024 + 512 + c] = var;
}

// ---------------------------------------------------------------------------
// Small dense FC: out[m,n] = act(in[m,:] @ w[:,n] + bias[n]) (+ resid)
// mode: 0=identity, 1=sigmoid, 2=PReLU(alpha)
// ---------------------------------------------------------------------------
__global__ void fc_kernel(const float* __restrict__ in, const float* __restrict__ w,
                          const float* __restrict__ bias, float* __restrict__ out,
                          int M, int K, int N, int mode,
                          const float* __restrict__ prelu_a,
                          const float* __restrict__ resid) {
  int idx = blockIdx.x * 256 + threadIdx.x;
  if (idx >= M * N) return;
  int m = idx / N;
  int n = idx - m * N;
  float acc = bias[n];
  const float* ip = in + (long)m * K;
  for (int k = 0; k < K; ++k) acc += ip[k] * w[(long)k * N + n];
  if (mode == 1) acc = sigf(acc);
  else if (mode == 2) { float a = prelu_a[0]; acc = (acc >= 0.0f) ? acc : a * acc; }
  if (resid) acc += resid[idx];
  out[idx] = acc;
}

// ---------------------------------------------------------------------------
// LayerNorm over 512 features, one row per block (512 threads).
// ---------------------------------------------------------------------------
__global__ __launch_bounds__(512) void ln_kernel(const float* __restrict__ in,
                                                 const float* __restrict__ g,
                                                 const float* __restrict__ b,
                                                 float* __restrict__ out) {
  __shared__ float sd[512];
  int row = blockIdx.x;
  int tid = threadIdx.x;
  float x = in[row * 512 + tid];
  sd[tid] = x;
  __syncthreads();
  for (int s = 256; s > 0; s >>= 1) {
    if (tid < s) sd[tid] += sd[tid + s];
    __syncthreads();
  }
  float mu = sd[0] * (1.0f / 512.0f);
  __syncthreads();
  float d = x - mu;
  sd[tid] = d * d;
  __syncthreads();
  for (int s = 256; s > 0; s >>= 1) {
    if (tid < s) sd[tid] += sd[tid + s];
    __syncthreads();
  }
  float var = sd[0] * (1.0f / 512.0f);
  out[row * 512 + tid] = d * rsqrtf(var + 1e-5f) * g[tid] + b[tid];
}

// ---------------------------------------------------------------------------
// Grouped head: out2[b,g] = sig( sum_f sig(in[b, g*4+f]) * wg[g,f] + bg[g] )
// ---------------------------------------------------------------------------
__global__ void grouped_kernel(const float* __restrict__ in, const float* __restrict__ wg,
                               const float* __restrict__ bg, float* __restrict__ out) {
  int idx = blockIdx.x * 256 + threadIdx.x;
  if (idx >= 64 * 128) return;
  int b = idx >> 7;
  int g = idx & 127;
  float acc = bg[g];
#pragma unroll
  for (int f = 0; f < 4; ++f) acc += sigf(in[b * 512 + g * 4 + f]) * wg[g * 4 + f];
  out[idx] = sigf(acc);
}

// ---------------------------------------------------------------------------
extern "C" void kernel_launch(void* const* d_in, const int* in_sizes, int n_in,
                              void* d_out, int out_size, void* d_ws, size_t ws_size,
                              hipStream_t stream) {
  (void)in_sizes; (void)n_in; (void)out_size; (void)ws_size;
  const float* inp = (const float*)d_in[0];
  // tdnn layer l (0..4): params at d_in[1+6l .. 1+6l+5] = {w, b, g, bb, m, v}
  auto P = [&](int l, int j) { return (const float*)d_in[1 + 6 * l + j]; };
  const float* w6  = (const float*)d_in[31]; const float* b6  = (const float*)d_in[32];
  const float* w7  = (const float*)d_in[33]; const float* b7  = (const float*)d_in[34];
  const float* wo  = (const float*)d_in[35]; const float* bo  = (const float*)d_in[36];
  const float* lng = (const float*)d_in[37]; const float* lnb = (const float*)d_in[38];
  const float* w11 = (const float*)d_in[39]; const float* b11 = (const float*)d_in[40];
  const float* pra = (const float*)d_in[41];
  const float* w12 = (const float*)d_in[42]; const float* b12 = (const float*)d_in[43];
  const float* w2  = (const float*)d_in[44]; const float* b2  = (const float*)d_in[45];
  const float* wg  = (const float*)d_in[46]; const float* bg  = (const float*)d_in[47];

  // Workspace layout
  char* ws = (char*)d_ws;
  __bf16* A0   = (__bf16*)ws; ws += (size_t)64 * 1000 * 64 * 2;    // padded input
  __bf16* ACTA = (__bf16*)ws; ws += (size_t)64 * 1000 * 512 * 2;   // ping
  __bf16* ACTB = (__bf16*)ws; ws += (size_t)64 * 1000 * 512 * 2;   // pong
  __bf16* WT   = (__bf16*)ws; ws += (size_t)512 * 1536 * 2;        // staged bf16 weights
  float* stat  = (float*)ws;  ws += (size_t)64 * 1024 * 4;
  float* t0    = (float*)ws;  ws += (size_t)64 * 512 * 4;
  float* t2    = (float*)ws;  ws += (size_t)64 * 512 * 4;
  float* t3    = (float*)ws;  ws += (size_t)64 * 512 * 4;

  // 1) pad input to 64 channels, bf16
  pad_input_kernel<<<(64 * 1000 * 64 + 255) / 256, 256, 0, stream>>>(inp, A0, 64 * 1000 * 64);

  // 2) five TDNN layers (stage weights, then WMMA GEMM). Stream order serializes
  //    reuse of WT. cfg: {Cin, CinPad, shift, dil, Tout, K}
  struct { int Cin, CinPad, shift, dil, Tout, K; const __bf16* in; __bf16* out; } cfg[5] = {
      {40, 64, 6, 1, 996, 5 * 64,   A0,   ACTA},
      {512, 512, 9, 1, 994, 3 * 512, ACTA, ACTB},
      {512, 512, 9, 2, 992, 2 * 512, ACTB, ACTA},
      {512, 512, 9, 1, 992, 512,     ACTA, ACTB},
      {512, 512, 9, 1, 992, 512,     ACTB, ACTA},
  };
  for (int l = 0; l < 5; ++l) {
    int K = cfg[l].K;
    wconv_kernel<<<(512 * K + 255) / 256, 256, 0, stream>>>(P(l, 0), WT, cfg[l].Cin,
                                                            cfg[l].CinPad, K);
    dim3 grid((64 * cfg[l].Tout) / 128, 8);
    tdnn_wmma_kernel<<<grid, 128, 0, stream>>>(cfg[l].in, WT, P(l, 1), P(l, 2), P(l, 3),
                                               P(l, 4), P(l, 5), cfg[l].out, cfg[l].Tout,
                                               cfg[l].shift, K, cfg[l].dil);
  }

  // 3) stats pooling over 992 frames -> (64, 1024)
  stats_kernel<<<64, 512, 0, stream>>>(ACTA, stat, 992);

  // 4) dense tail. x_vec goes straight into its d_out slice.
  float* out2 = (float*)d_out;          // (64,128)
  float* xvec = (float*)d_out + 8192;   // (64,512)
  fc_kernel<<<128, 256, 0, stream>>>(stat, w6, b6, t0, 64, 1024, 512, 0, nullptr, nullptr);
  fc_kernel<<<128, 256, 0, stream>>>(t0, w7, b7, xvec, 64, 512, 512, 0, nullptr, nullptr);
  fc_kernel<<<128, 256, 0, stream>>>(xvec, wo, bo, t2, 64, 512, 512, 1, nullptr, nullptr);
  ln_kernel<<<64, 512, 0, stream>>>(t2, lng, lnb, t3);
  fc_kernel<<<128, 256, 0, stream>>>(t3, w11, b11, t0, 64, 512, 512, 2, pra, nullptr);
  fc_kernel<<<128, 256, 0, stream>>>(t0, w12, b12, t3, 64, 512, 512, 1, nullptr, t2);
  fc_kernel<<<128, 256, 0, stream>>>(t3, w2, b2, t0, 64, 512, 512, 1, nullptr, nullptr);
  grouped_kernel<<<32, 256, 0, stream>>>(t0, wg, bg, out2);
}
